// Zeus_46428596470384
// MI455X (gfx1250) — matrix operand order repaired and checked by reading, not compiled, not run on hardware
//
#include <hip/hip_runtime.h>
#include <hip/hip_bf16.h>

// ---------------------------------------------------------------------------
// Problem constants (fixed by the reference setup)
// ---------------------------------------------------------------------------
#define BB 4
#define SS 4096
#define DD 2048          // D
#define FF 4096          // 2D
#define ROT_DIM 1024     // D/2 channels rotated
// log2(10000)/1024
#define FREQ_C 0.0129762816206537f

// Tile config for the fused GEMM kernel
#define MT 128           // rows (s) per workgroup
#define NT 64            // K-columns per workgroup (plus 64 paired V-columns)
#define KT 32            // K-depth per WMMA step
#define NKT (DD / KT)    // 64 K-steps
#define LDA_S 40         // LDS stride (halves) for A tile rows  (pad vs 32)
#define LDB_S 40         // LDS stride (halves) for B tile cols
#define LDC_S 132        // LDS stride (floats) for C tile rows  (pad vs 128)
#define ABUF (MT * LDA_S)        // halves per A buffer
#define BBUF (2 * NT * LDB_S)    // halves per B buffer

typedef __attribute__((ext_vector_type(16))) __bf16 v16bf;
typedef __attribute__((ext_vector_type(8)))  float  v8f;
typedef __attribute__((ext_vector_type(4)))  unsigned int u32x4;
typedef __attribute__((ext_vector_type(2)))  unsigned int u32x2;

union ABPack { v16bf v; u32x4 q[2]; };

__device__ __forceinline__ unsigned int f2bf_rne(float f) {
    unsigned int u = __float_as_uint(f);
    u += 0x7FFFu + ((u >> 16) & 1u);
    return u >> 16;
}
__device__ __forceinline__ unsigned int pack_bf2(float lo, float hi) {
    return f2bf_rne(lo) | (f2bf_rne(hi) << 16);
}

// ---------------------------------------------------------------------------
// Kernel 0: zero the A accumulator workspace (B*D floats)
// ---------------------------------------------------------------------------
__global__ void zeus_zeroA(float* __restrict__ Aws) {
    int i = blockIdx.x * blockDim.x + threadIdx.x;
    if (i < BB * DD) Aws[i] = 0.0f;
}

// ---------------------------------------------------------------------------
// Kernel 1: fused  KV = X@W + b  ->  rotary  ->  cos  ->  sum over s
// Grid: (DD/NT, SS/MT, BB), 256 threads (8 wave32).
// Double-buffered LDS pipeline: global float4 loads for tile t+1 are issued
// before the 8 back-to-back v_wmma_f32_16x16x32_bf16 on tile t.
// ---------------------------------------------------------------------------
__launch_bounds__(256)
__global__ void zeus_gemm_rot_reduce(const float* __restrict__ X,
                                     const float* __restrict__ W,
                                     const float* __restrict__ bias,
                                     float* __restrict__ Aws) {
    __shared__ unsigned short ldsA[2 * ABUF];   // X tile, bf16, [m][k], 2 bufs
    __shared__ unsigned short ldsB[2 * BBUF];   // W tile, bf16, [c][k] (transposed)
    __shared__ float          ldsC[MT * LDC_S]; // KV tile f32, then cos() values
    __shared__ float          ldsR[4 * NT];     // reduction scratch

    const int tid  = threadIdx.x;
    const int lane = tid & 31;
    const int wave = tid >> 5;                 // 0..7 -> M strip
    const int b    = blockIdx.z;
    const int s0   = blockIdx.y * MT;
    const int n0   = blockIdx.x * NT;

    const float* Xb = X + (size_t)b * SS * DD;

    // staging coordinates (each thread moves 4 float4 per tile per operand)
    const int a_kq = tid & 7;                  // float4 index within X row
    const int a_m0 = tid >> 3;                 // row, +32 per i
    const int b_c0 = (tid & 31) * 4;           // column group (0..124)
    const int b_k0 = tid >> 5;                 // k row, +8 per i
    const int b_wcol = (b_c0 < NT) ? (n0 + b_c0) : (DD + n0 + (b_c0 - NT));

    v8f acc[8];
#pragma unroll
    for (int nt = 0; nt < 8; ++nt) acc[nt] = (v8f)0.0f;

    // WMMA operand addressing (ISA 16-bit A 16x32 / B 32x16 layouts)
    const int am = lane & 15;
    const int ah = lane >> 4;
    const int aOff0 = (wave * 16 + am) * LDA_S + ah * 8;   // K = 8h..8h+7
    const int aOff1 = aOff0 + 16;                          // K = 16+8h..

    float4 ra[4], rb[4];

    // ---- tile loaders / storer ----------------------------------------
    auto loadTile = [&](int kt) {
        const int k0 = kt * KT;
#pragma unroll
        for (int i = 0; i < 4; ++i)
            ra[i] = *(const float4*)(Xb + (size_t)(s0 + a_m0 + i * 32) * DD + k0 + a_kq * 4);
#pragma unroll
        for (int i = 0; i < 4; ++i)
            rb[i] = *(const float4*)(W + (size_t)(k0 + b_k0 + i * 8) * FF + b_wcol);
    };
    auto storeTile = [&](int p) {
        unsigned short* A = ldsA + p * ABUF;
        unsigned short* B = ldsB + p * BBUF;
#pragma unroll
        for (int i = 0; i < 4; ++i) {
            u32x2 w2 = { pack_bf2(ra[i].x, ra[i].y), pack_bf2(ra[i].z, ra[i].w) };
            *(u32x2*)(A + (a_m0 + i * 32) * LDA_S + a_kq * 4) = w2;   // ds_store_b64
        }
#pragma unroll
        for (int i = 0; i < 4; ++i) {
            const int k = b_k0 + i * 8;
            B[(b_c0 + 0) * LDB_S + k] = (unsigned short)f2bf_rne(rb[i].x);
            B[(b_c0 + 1) * LDB_S + k] = (unsigned short)f2bf_rne(rb[i].y);
            B[(b_c0 + 2) * LDB_S + k] = (unsigned short)f2bf_rne(rb[i].z);
            B[(b_c0 + 3) * LDB_S + k] = (unsigned short)f2bf_rne(rb[i].w);
        }
    };

    // ---- pipelined GEMM main loop --------------------------------------
    loadTile(0);
    storeTile(0);
    __syncthreads();

    int p = 0;
    for (int kt = 0; kt < NKT; ++kt) {
        const bool more = (kt + 1 < NKT);
        if (more) loadTile(kt + 1);            // global loads in flight

        const unsigned short* A  = ldsA + p * ABUF;
        const unsigned short* Bp = ldsB + p * BBUF;

        ABPack pa;
        pa.q[0] = *(const u32x4*)(A + aOff0);
        pa.q[1] = *(const u32x4*)(A + aOff1);

        ABPack pb[8];
#pragma unroll
        for (int nt = 0; nt < 8; ++nt) {
            const unsigned short* bP = Bp + (nt * 16 + am) * LDB_S + ah * 8;
            pb[nt].q[0] = *(const u32x4*)bP;
            pb[nt].q[1] = *(const u32x4*)(bP + 16);
        }
#pragma unroll
        for (int nt = 0; nt < 8; ++nt)
            acc[nt] = __builtin_amdgcn_wmma_f32_16x16x32_bf16(
                false, pa.v, false, pb[nt].v, (short)0, acc[nt], false, false);

        if (more) storeTile(p ^ 1);
        __syncthreads();
        p ^= 1;
    }

    // ---- spill accumulators to LDS (C/D layout: VGPR r -> M = r+8*(lane>>4),
    //      N = lane&15) ----
    const int rowBase = wave * 16 + (lane >> 4) * 8;
    const int colB = lane & 15;
#pragma unroll
    for (int nt = 0; nt < 8; ++nt) {
#pragma unroll
        for (int r = 0; r < 8; ++r)
            ldsC[(rowBase + r) * LDC_S + nt * 16 + colB] = acc[nt][r];
    }
    __syncthreads();

    // ---- rotary (+bias) + cos, in place. Each thread owns fixed column pair
    //      (c0,c1) of either the K half or the V half, loops over rows. ----
    {
        const int jj   = tid & 31;              // pair index within 64 columns
        const int part = (tid >> 5) & 1;        // 0 = K half, 1 = V half
        const int c0 = part * NT + 2 * jj;
        const int c1 = c0 + 1;
        const int ch = n0 + 2 * jj;             // rotary channel (even)
        const float b0 = bias[(part ? DD : 0) + n0 + 2 * jj];
        const float b1 = bias[(part ? DD : 0) + n0 + 2 * jj + 1];
        const bool rot = (ch < ROT_DIM);
        const float invf = rot ? exp2f(-(float)ch * FREQ_C) : 0.0f;
        const int mBase = tid >> 6;             // 0..3
#pragma unroll 4
        for (int i = 0; i < 32; ++i) {
            const int m = mBase + 4 * i;
            float v0 = ldsC[m * LDC_S + c0] + b0;
            float v1 = ldsC[m * LDC_S + c1] + b1;
            float r0 = v0, r1 = v1;
            if (rot) {
                float th = (float)(s0 + m) * invf;
                float sn, cs;
                __sincosf(th, &sn, &cs);
                r0 = v0 * cs - v1 * sn;
                r1 = v1 * cs + v0 * sn;
            }
            ldsC[m * LDC_S + c0] = __cosf(r0);
            ldsC[m * LDC_S + c1] = __cosf(r1);
        }
    }
    __syncthreads();

    // ---- reduce cos(K)*cos(V) over the 128 rows, one atomic per column ----
    {
        const int pcol = tid & 63;
        const int qr   = tid >> 6;
        float partial = 0.0f;
#pragma unroll 8
        for (int m = qr * 32; m < qr * 32 + 32; ++m)
            partial += ldsC[m * LDC_S + pcol] * ldsC[m * LDC_S + NT + pcol];
        ldsR[qr * NT + pcol] = partial;
    }
    __syncthreads();
    if (tid < NT) {
        float tot = ldsR[tid] + ldsR[NT + tid] + ldsR[2 * NT + tid] + ldsR[3 * NT + tid];
        atomicAdd(Aws + ((size_t)b * DD) + n0 + tid, tot);
    }
}

// ---------------------------------------------------------------------------
// Kernel 2: out[b,s,d] = (0.5*A[b,d] + 0.5) * X[b,s,d]   (float4 streaming)
// ---------------------------------------------------------------------------
__launch_bounds__(256)
__global__ void zeus_finalize(const float* __restrict__ X,
                              const float* __restrict__ Aws,
                              float* __restrict__ out) {
    size_t i = (size_t)blockIdx.x * blockDim.x + threadIdx.x;   // float4 idx
    size_t flat = i * 4;
    int d = (int)(flat & (DD - 1));
    int b = (int)(flat >> 23);                                  // S*D = 2^23
    float4 x = ((const float4*)X)[i];
    float4 a = *(const float4*)(Aws + ((size_t)b * DD) + d);
    float4 o;
    o.x = (0.5f * a.x + 0.5f) * x.x;
    o.y = (0.5f * a.y + 0.5f) * x.y;
    o.z = (0.5f * a.z + 0.5f) * x.z;
    o.w = (0.5f * a.w + 0.5f) * x.w;
    ((float4*)out)[i] = o;
}

// ---------------------------------------------------------------------------
extern "C" void kernel_launch(void* const* d_in, const int* in_sizes, int n_in,
                              void* d_out, int out_size, void* d_ws, size_t ws_size,
                              hipStream_t stream) {
    const float* X = (const float*)d_in[0];
    const float* W = (const float*)d_in[1];
    const float* bia = (const float*)d_in[2];
    float* out = (float*)d_out;
    float* Aws = (float*)d_ws;                  // B*D floats = 32 KB

    (void)in_sizes; (void)n_in; (void)out_size; (void)ws_size;

    zeus_zeroA<<<(BB * DD + 255) / 256, 256, 0, stream>>>(Aws);

    dim3 grid1(DD / NT, SS / MT, BB);           // (32, 32, 4)
    zeus_gemm_rot_reduce<<<grid1, 256, 0, stream>>>(X, W, bia, Aws);

    size_t nq = (size_t)BB * SS * DD / 4;       // float4 count
    zeus_finalize<<<(unsigned)(nq / 256), 256, 0, stream>>>(X, Aws, out);
}
